// MultiHeadAttention_4793183502697
// MI455X (gfx1250) — matrix operand order repaired and checked by reading, not compiled, not run on hardware
//
#include <hip/hip_runtime.h>
#include <math.h>

#define D_MODEL 1024
#define NHEAD   16
#define HD      64
#define BATCH   4
#define SEQ     2048
#define MTOT    (BATCH * SEQ)

typedef __attribute__((ext_vector_type(16))) __bf16 bf16x16;
typedef __attribute__((ext_vector_type(8)))  __bf16 bf16x8;
typedef __attribute__((ext_vector_type(8)))  float  f32x8;

__device__ __forceinline__ f32x8 wmma_bf16(bf16x16 a, bf16x16 b, f32x8 c) {
    return __builtin_amdgcn_wmma_f32_16x16x32_bf16(false, a, false, b, (short)0, c, false, false);
}

// A-matrix 16x32 bf16 fragment: lane (row, half g) holds contiguous K
// [16g, 16g+16), matching the B-fragment order.
__device__ __forceinline__ bf16x16 load_afrag(const __bf16* p, int g) {
    bf16x8 c0 = *(const bf16x8*)(p + 16 * g);
    bf16x8 c1 = *(const bf16x8*)(p + 16 * g + 8);
    bf16x16 r;
#pragma unroll
    for (int i = 0; i < 8; ++i) { r[i] = c0[i]; r[i + 8] = c1[i]; }
    return r;
}

// B-matrix 32x16 bf16 fragment: lane (col, half g) holds contiguous K [16g, 16g+16).
__device__ __forceinline__ bf16x16 load_bfrag(const __bf16* p, int g) {
    return *(const bf16x16*)(p + 16 * g);
}

__device__ __forceinline__ __bf16 bf_hi(float x) { return (__bf16)x; }
__device__ __forceinline__ __bf16 bf_lo(float x, __bf16 h) { return (__bf16)(x - (float)h); }

// ---------------- elementwise fp32 -> bf16 hi/lo split ----------------
__global__ void split_kernel(const float* __restrict__ in,
                             __bf16* __restrict__ hi, __bf16* __restrict__ lo, int n) {
    int i = blockIdx.x * blockDim.x + threadIdx.x;
    if (i < n) {
        float x = in[i];
        __bf16 h = bf_hi(x);
        hi[i] = h;
        lo[i] = bf_lo(x, h);
    }
}

// ---------------- weight transpose + split: W[K][N] -> Wt[N][K] hi/lo ----------------
__global__ void wsplit_kernel(const float* __restrict__ w,
                              __bf16* __restrict__ thi, __bf16* __restrict__ tlo) {
    int i = blockIdx.x * blockDim.x + threadIdx.x;   // output index n*K + k
    int n = i >> 10, k = i & (D_MODEL - 1);
    float x = w[(size_t)k * D_MODEL + n];
    __bf16 h = bf_hi(x);
    thi[i] = h;
    tlo[i] = bf_lo(x, h);
}

// ---------------- GEMM: C[M x N] = A[M x K] * Bt[N x K]^T + bias ----------------
// A, Bt given as bf16 hi/lo pairs; 3-term WMMA per k-chunk for ~fp32 accuracy.
// Outputs: outF fp32 row-major, or outHi/outLo bf16 split (vtrans=1 stores the
// per-head transposed V^T layout [(b*H+h)*HD + d][S] for the attention PV GEMM).
__global__ __launch_bounds__(256)
void gemm_kernel(const __bf16* __restrict__ Ahi, const __bf16* __restrict__ Alo,
                 const __bf16* __restrict__ Bthi, const __bf16* __restrict__ Btlo,
                 const float* __restrict__ bias,
                 float* __restrict__ outF,
                 __bf16* __restrict__ outHi, __bf16* __restrict__ outLo,
                 int vtrans) {
    const int K = D_MODEL, N = D_MODEL;
    const int lane = threadIdx.x & 31, w = threadIdx.x >> 5;
    const int lq = lane & 15, g = lane >> 4;
    const int wm = w & 3, wn = w >> 2;
    const int mBase = blockIdx.y * 128 + wm * 32;
    const int nBase = blockIdx.x * 128 + wn * 64;

    const f32x8 zero = {0.f, 0.f, 0.f, 0.f, 0.f, 0.f, 0.f, 0.f};
    f32x8 acc[2][4];
#pragma unroll
    for (int mt = 0; mt < 2; ++mt)
#pragma unroll
        for (int nt = 0; nt < 4; ++nt) acc[mt][nt] = zero;

    for (int kk = 0; kk < K; kk += 32) {
        bf16x16 ah[2], al[2];
#pragma unroll
        for (int mt = 0; mt < 2; ++mt) {
            size_t ra = (size_t)(mBase + mt * 16 + lq) * K + kk;
            ah[mt] = load_afrag(Ahi + ra, g);
            al[mt] = load_afrag(Alo + ra, g);
        }
#pragma unroll
        for (int nt = 0; nt < 4; ++nt) {
            size_t rb = (size_t)(nBase + nt * 16 + lq) * K + kk;
            bf16x16 bh = load_bfrag(Bthi + rb, g);
            bf16x16 bl = load_bfrag(Btlo + rb, g);
#pragma unroll
            for (int mt = 0; mt < 2; ++mt) {
                acc[mt][nt] = wmma_bf16(ah[mt], bh, acc[mt][nt]);
                acc[mt][nt] = wmma_bf16(ah[mt], bl, acc[mt][nt]);
                acc[mt][nt] = wmma_bf16(al[mt], bh, acc[mt][nt]);
            }
        }
    }

#pragma unroll
    for (int mt = 0; mt < 2; ++mt)
#pragma unroll
        for (int nt = 0; nt < 4; ++nt) {
            const int n = nBase + nt * 16 + lq;          // C layout: col = lane%16
            const float bv = bias[n];
            const int m0 = mBase + mt * 16 + 8 * g;      // C layout: row = j + 8*(lane/16)
            if (outF) {
#pragma unroll
                for (int j = 0; j < 8; ++j)
                    outF[(size_t)(m0 + j) * N + n] = acc[mt][nt][j] + bv;
            }
            if (outHi) {
                if (!vtrans) {
#pragma unroll
                    for (int j = 0; j < 8; ++j) {
                        float v = acc[mt][nt][j] + bv;
                        __bf16 h = bf_hi(v);
                        outHi[(size_t)(m0 + j) * N + n] = h;
                        outLo[(size_t)(m0 + j) * N + n] = bf_lo(v, h);
                    }
                } else {
                    const int b  = m0 >> 11, s0 = m0 & (SEQ - 1);
                    const int hh = n >> 6,  dc = n & (HD - 1);
                    const size_t base = (((size_t)b * NHEAD + hh) * HD + dc) * SEQ + s0;
                    bf16x8 hv, lv;
#pragma unroll
                    for (int j = 0; j < 8; ++j) {
                        float v = acc[mt][nt][j] + bv;
                        __bf16 h = bf_hi(v);
                        hv[j] = h;
                        lv[j] = bf_lo(v, h);
                    }
                    *(bf16x8*)(outHi + base) = hv;   // 8 consecutive seq positions
                    *(bf16x8*)(outLo + base) = lv;
                }
            }
        }
}

// ---------------- Flash attention, computed transposed (S^T = K*Q^T, O^T = V^T*P^T) ----
// 4 waves / block, each wave owns 16 query rows. No LDS: softmax stats live in a
// lane pair (col q = lane%16), P reaches B-operand layout via shfl_xor(.,16).
__global__ __launch_bounds__(128)
void attn_kernel(const __bf16* __restrict__ qhi, const __bf16* __restrict__ qlo,
                 const __bf16* __restrict__ khi, const __bf16* __restrict__ klo,
                 const __bf16* __restrict__ vthi, const __bf16* __restrict__ vtlo,
                 __bf16* __restrict__ chi, __bf16* __restrict__ clo) {
    const int lane = threadIdx.x & 31, w = threadIdx.x >> 5;
    const int lq = lane & 15, g = lane >> 4;
    const int bh = blockIdx.y, b = bh >> 4, h = bh & 15;
    const int qRow0 = blockIdx.x * 64 + w * 16;

    const size_t qkBase = ((size_t)b * SEQ) * D_MODEL + h * HD;
    const size_t vBase  = (((size_t)b * NHEAD + h) * HD) * SEQ;

    // Q fragments (B operand of S^T): lane = query col, contiguous d.
    bf16x16 bq_h[2], bq_l[2];
#pragma unroll
    for (int kt = 0; kt < 2; ++kt) {
        size_t off = qkBase + (size_t)(qRow0 + lq) * D_MODEL + kt * 32 + 16 * g;
        bq_h[kt] = *(const bf16x16*)(qhi + off);
        bq_l[kt] = *(const bf16x16*)(qlo + off);
    }

    const f32x8 zero = {0.f, 0.f, 0.f, 0.f, 0.f, 0.f, 0.f, 0.f};
    f32x8 o[4];
#pragma unroll
    for (int dt = 0; dt < 4; ++dt) o[dt] = zero;
    float mstat = -__builtin_inff(), lstat = 0.0f;

    const int kbLast = (qRow0 + 15) >> 6;
    for (int kb = 0; kb <= kbLast; ++kb) {
        const int kBase = kb * 64;
        f32x8 s[4];
#pragma unroll
        for (int nt = 0; nt < 4; ++nt) s[nt] = zero;

        // S^T accumulate: A = K tile (rows = keys), B = Q^T
#pragma unroll
        for (int nt = 0; nt < 4; ++nt) {
            const size_t rk = qkBase + (size_t)(kBase + nt * 16 + lq) * D_MODEL;
#pragma unroll
            for (int kt = 0; kt < 2; ++kt) {
                bf16x16 akh = load_afrag(khi + rk + kt * 32, g);
                bf16x16 akl = load_afrag(klo + rk + kt * 32, g);
                s[nt] = wmma_bf16(akh, bq_h[kt], s[nt]);
                s[nt] = wmma_bf16(akh, bq_l[kt], s[nt]);
                s[nt] = wmma_bf16(akl, bq_h[kt], s[nt]);
            }
        }

        // scale + causal mask (cndmask only; EXEC stays full for WMMA)
        float bm = -__builtin_inff();
#pragma unroll
        for (int nt = 0; nt < 4; ++nt)
#pragma unroll
            for (int j = 0; j < 8; ++j) {
                const int key = kBase + nt * 16 + 8 * g + j;
                float v = s[nt][j] * 0.125f;
                v = (key <= qRow0 + lq) ? v : -__builtin_inff();
                s[nt][j] = v;
                bm = fmaxf(bm, v);
            }
        bm = fmaxf(bm, __shfl_xor(bm, 16, 32));   // combine lane pair (full column)
        const float mnew  = fmaxf(mstat, bm);
        const float alpha = __expf(mstat - mnew);
        mstat = mnew;

        float rs = 0.0f;
#pragma unroll
        for (int nt = 0; nt < 4; ++nt)
#pragma unroll
            for (int j = 0; j < 8; ++j) {
                const float p = __expf(s[nt][j] - mnew);
                s[nt][j] = p;
                rs += p;
            }
        rs += __shfl_xor(rs, 16, 32);
        lstat = lstat * alpha + rs;
#pragma unroll
        for (int dt = 0; dt < 4; ++dt)
#pragma unroll
            for (int j = 0; j < 8; ++j) o[dt][j] *= alpha;

        // O^T += V^T * P^T.  Build P^T B-fragments from C-layout via xor-16 exchange.
#pragma unroll
        for (int kt2 = 0; kt2 < 2; ++kt2) {
            const int t0 = kt2 * 2, t1 = t0 + 1;
            bf16x16 ph, pl;
#pragma unroll
            for (int j = 0; j < 8; ++j) {
                const float a0 = s[t0][j], a1 = s[t1][j];
                const float p0 = __shfl_xor(a0, 16, 32);
                const float p1 = __shfl_xor(a1, 16, 32);
                const float e0 = g ? p1 : a0;   // K = 16g + j
                const float e1 = g ? a1 : p0;   // K = 16g + 8 + j
                __bf16 h0 = bf_hi(e0);
                __bf16 h1 = bf_hi(e1);
                ph[j]     = h0;
                pl[j]     = bf_lo(e0, h0);
                ph[8 + j] = h1;
                pl[8 + j] = bf_lo(e1, h1);
            }
#pragma unroll
            for (int dt = 0; dt < 4; ++dt) {
                const size_t rv = vBase + (size_t)(dt * 16 + lq) * SEQ + kBase + kt2 * 32;
                bf16x16 vh = load_afrag(vthi + rv, g);
                bf16x16 vl = load_afrag(vtlo + rv, g);
                o[dt] = wmma_bf16(vh, ph, o[dt]);
                o[dt] = wmma_bf16(vh, pl, o[dt]);
                o[dt] = wmma_bf16(vl, ph, o[dt]);
            }
        }
    }

    // Normalize, split, store ctx in [B,S,D] hi/lo (A-operand layout for final GEMM).
    const float rinv = 1.0f / lstat;
    const size_t crow = ((size_t)b * SEQ + qRow0 + lq) * D_MODEL + h * HD;
#pragma unroll
    for (int dt = 0; dt < 4; ++dt) {
        bf16x8 hv, lv;
#pragma unroll
        for (int j = 0; j < 8; ++j) {
            float v = o[dt][j] * rinv;
            __bf16 h = bf_hi(v);
            hv[j] = h;
            lv[j] = bf_lo(v, h);
        }
        const size_t off = crow + dt * 16 + 8 * g;
        *(bf16x8*)(chi + off) = hv;
        *(bf16x8*)(clo + off) = lv;
    }
}

extern "C" void kernel_launch(void* const* d_in, const int* in_sizes, int n_in,
                              void* d_out, int out_size, void* d_ws, size_t ws_size,
                              hipStream_t stream) {
    (void)in_sizes; (void)n_in; (void)out_size; (void)ws_size;
    const float* X  = (const float*)d_in[0];
    const float* Wq = (const float*)d_in[1];
    const float* bq = (const float*)d_in[2];
    const float* Wk = (const float*)d_in[3];
    const float* bk = (const float*)d_in[4];
    const float* Wv = (const float*)d_in[5];
    const float* bv = (const float*)d_in[6];
    const float* Wo = (const float*)d_in[7];
    const float* bo = (const float*)d_in[8];

    char* ws = (char*)d_ws;
    size_t off = 0;
    auto take = [&](size_t bytes) -> char* {
        char* p = ws + off;
        off += (bytes + 255) & ~(size_t)255;
        return p;
    };
    const size_t actB = (size_t)MTOT * D_MODEL * sizeof(__bf16);  // 16 MB
    const size_t wB   = (size_t)D_MODEL * D_MODEL * sizeof(__bf16);

    __bf16* x_hi  = (__bf16*)take(actB); __bf16* x_lo  = (__bf16*)take(actB);
    __bf16* wqh   = (__bf16*)take(wB);   __bf16* wql   = (__bf16*)take(wB);
    __bf16* wkh   = (__bf16*)take(wB);   __bf16* wkl   = (__bf16*)take(wB);
    __bf16* wvh   = (__bf16*)take(wB);   __bf16* wvl   = (__bf16*)take(wB);
    __bf16* woh   = (__bf16*)take(wB);   __bf16* wol   = (__bf16*)take(wB);
    __bf16* q_hi  = (__bf16*)take(actB); __bf16* q_lo  = (__bf16*)take(actB);
    __bf16* k_hi  = (__bf16*)take(actB); __bf16* k_lo  = (__bf16*)take(actB);
    __bf16* vt_hi = (__bf16*)take(actB); __bf16* vt_lo = (__bf16*)take(actB);
    __bf16* c_hi  = (__bf16*)take(actB); __bf16* c_lo  = (__bf16*)take(actB);

    const int nAct = MTOT * D_MODEL;          // 8M
    const int nW   = D_MODEL * D_MODEL;       // 1M
    split_kernel<<<nAct / 256, 256, 0, stream>>>(X, x_hi, x_lo, nAct);
    wsplit_kernel<<<nW / 256, 256, 0, stream>>>(Wq, wqh, wql);
    wsplit_kernel<<<nW / 256, 256, 0, stream>>>(Wk, wkh, wkl);
    wsplit_kernel<<<nW / 256, 256, 0, stream>>>(Wv, wvh, wvl);
    wsplit_kernel<<<nW / 256, 256, 0, stream>>>(Wo, woh, wol);

    dim3 ggrid(D_MODEL / 128, MTOT / 128);    // (8, 64)
    gemm_kernel<<<ggrid, 256, 0, stream>>>(x_hi, x_lo, wqh, wql, bq,
                                           nullptr, q_hi, q_lo, 0);
    gemm_kernel<<<ggrid, 256, 0, stream>>>(x_hi, x_lo, wkh, wkl, bk,
                                           nullptr, k_hi, k_lo, 0);
    gemm_kernel<<<ggrid, 256, 0, stream>>>(x_hi, x_lo, wvh, wvl, bv,
                                           nullptr, vt_hi, vt_lo, 1);

    dim3 agrid(SEQ / 64, BATCH * NHEAD);      // (32, 64)
    attn_kernel<<<agrid, 128, 0, stream>>>(q_hi, q_lo, k_hi, k_lo,
                                           vt_hi, vt_lo, c_hi, c_lo);

    gemm_kernel<<<ggrid, 256, 0, stream>>>(c_hi, c_lo, woh, wol, bo,
                                           (float*)d_out, nullptr, nullptr, 0);
}
